// MultiHeadAttention_40149354283071
// MI455X (gfx1250) — compile-verified, matrix operand order checked by reading, not run on hardware
//
#include <hip/hip_runtime.h>
#include <hip/hip_bf16.h>
#include <cstdint>
#include <cstddef>

typedef __bf16 bf16_t;
typedef __attribute__((ext_vector_type(16))) __bf16 v16bf;
typedef __attribute__((ext_vector_type(4)))  __bf16 v4bf;
typedef __attribute__((ext_vector_type(8)))  float  v8f;
typedef __attribute__((ext_vector_type(4)))  int    v4i;
typedef __attribute__((ext_vector_type(8)))  int    v8i;

#define S_LEN  2048
#define HID    2048
#define NHEADS 16
#define HDIM   128
#define BATCH  2
#define MROWS  (BATCH * S_LEN)   // 4096

// ---------------------------------------------------------------------------
// WMMA + async-LDS helpers
// ---------------------------------------------------------------------------
__device__ __forceinline__ v8f wmma_bf16(v16bf a, v16bf b, v8f c) {
  // D = A(16x32 bf16) * B(32x16 bf16, stored as [N,K] rows like A) + C(f32)
  return __builtin_amdgcn_wmma_f32_16x16x32_bf16(
      /*neg_a=*/false, a, /*neg_b=*/false, b,
      /*c_mod=*/(short)0, c, /*reuse_a=*/false, /*reuse_b=*/false);
}

// Load one 16x32 bf16 A/B fragment from row-major [row, K] storage.
// Lane L<16: row L, halfs k=0..7 and 16..23; lane L>=16: row L-16, k=8..15, 24..31.
__device__ __forceinline__ v16bf load_frag(const bf16_t* base, int ld_halfs) {
  const int lane = threadIdx.x & 31;
  const bf16_t* p = base + (lane & 15) * ld_halfs + ((lane >> 4) << 3);
  v4i lo = *(const v4i*)(p);
  v4i hi = *(const v4i*)(p + 16);
  return __builtin_bit_cast(v16bf,
         __builtin_shufflevector(lo, hi, 0, 1, 2, 3, 4, 5, 6, 7));
}

// Per-lane async DMA: 16B global -> LDS, tracked on ASYNCcnt (no data VGPRs).
// LDS address = low 32 bits of generic pointer.
__device__ __forceinline__ void async_copy16(const bf16_t* g, bf16_t* l) {
  unsigned lds_off = (unsigned)(uintptr_t)l;
  asm volatile("global_load_async_to_lds_b128 %0, %1, off"
               :: "v"(lds_off), "v"(g) : "memory");
}
__device__ __forceinline__ void wait_async() {
  asm volatile("s_wait_asynccnt 0x0" ::: "memory");
}

// ---------------------------------------------------------------------------
// fp32 -> bf16 conversion (vectorized x4)
// ---------------------------------------------------------------------------
__global__ void cvt_f32_to_bf16(const float* __restrict__ in,
                                bf16_t* __restrict__ out, int n4) {
  int i = blockIdx.x * blockDim.x + threadIdx.x;
  if (i >= n4) return;
  float4 v = ((const float4*)in)[i];
  v4bf o = { (bf16_t)v.x, (bf16_t)v.y, (bf16_t)v.z, (bf16_t)v.w };
  ((v4bf*)out)[i] = o;
}

// ---------------------------------------------------------------------------
// C = A[M,K] * B[N,K]^T  (both bf16 row-major), fp32 accumulate.
// Block tile 128x128, 8 waves (4x2), wave tile 32x64, K-step 32.
// Double-buffered LDS with async global->LDS staging overlapping WMMA.
// ---------------------------------------------------------------------------
#define GTM 128
#define GTN 128
#define GKS 32
#define GLDA 40   // padded LDS row stride in halfs (80B: 16B aligned, conflict-free)

template <bool OUT_BF16>
__global__ __launch_bounds__(256)
void gemm_bt(const bf16_t* __restrict__ A, const bf16_t* __restrict__ B,
             float* __restrict__ Cf, bf16_t* __restrict__ Cb,
             int M, int N, int K) {
  __shared__ __attribute__((aligned(16))) bf16_t sA[2][GTM * GLDA];
  __shared__ __attribute__((aligned(16))) bf16_t sB[2][GTN * GLDA];
  const int tid = threadIdx.x, wave = tid >> 5, lane = tid & 31;
  const int mblk = blockIdx.y * GTM, nblk = blockIdx.x * GTN;
  const int wm = (wave >> 1) * 32, wn = (wave & 1) * 64;

  auto stage = [&](int k0, int buf) {
#pragma unroll
    for (int s = 0; s < 2; ++s) {
      int seg = tid + s * 256;                 // 0..511
      int row = seg >> 2, c = (seg & 3) << 3;  // 128 rows x 4 x 16B segments
      async_copy16(A + (size_t)(mblk + row) * K + k0 + c,
                   &sA[buf][row * GLDA + c]);
      async_copy16(B + (size_t)(nblk + row) * K + k0 + c,
                   &sB[buf][row * GLDA + c]);
    }
  };

  const v8f zero8 = {0.f, 0.f, 0.f, 0.f, 0.f, 0.f, 0.f, 0.f};
  v8f acc[2][4];
#pragma unroll
  for (int i = 0; i < 2; ++i)
#pragma unroll
    for (int j = 0; j < 4; ++j) acc[i][j] = zero8;

  stage(0, 0);
  wait_async();
  __syncthreads();

  int buf = 0;
  for (int k0 = 0; k0 < K; k0 += GKS) {
    if (k0 + GKS < K) stage(k0 + GKS, buf ^ 1);  // async prefetch next tile

    v16bf af[2], bfr[4];
#pragma unroll
    for (int i = 0; i < 2; ++i)
      af[i] = load_frag(&sA[buf][(wm + i * 16) * GLDA], GLDA);
#pragma unroll
    for (int j = 0; j < 4; ++j)
      bfr[j] = load_frag(&sB[buf][(wn + j * 16) * GLDA], GLDA);
#pragma unroll
    for (int i = 0; i < 2; ++i)
#pragma unroll
      for (int j = 0; j < 4; ++j) acc[i][j] = wmma_bf16(af[i], bfr[j], acc[i][j]);

    wait_async();      // our async copies into buf^1 are done
    __syncthreads();   // everyone's are done; buf free for next prefetch
    buf ^= 1;
  }

  // C layout: VGPR r -> row r (lanes 0-15) / row r+8 (lanes 16-31); lane&15 -> col
  const int coll = lane & 15, rtop = (lane >> 4) << 3;
#pragma unroll
  for (int i = 0; i < 2; ++i)
#pragma unroll
    for (int j = 0; j < 4; ++j)
#pragma unroll
      for (int r = 0; r < 8; ++r) {
        size_t row = (size_t)(mblk + wm + i * 16 + rtop + r);
        size_t col = (size_t)(nblk + wn + j * 16 + coll);
        if (OUT_BF16) Cb[row * N + col] = (bf16_t)acc[i][j][r];
        else          Cf[row * N + col] = acc[i][j][r];
      }
}

// ---------------------------------------------------------------------------
// RoPE on Q and K (in-place, bf16 storage, f32 math). One thread per (row,h,pair).
// ---------------------------------------------------------------------------
__global__ void rope_kernel(bf16_t* __restrict__ Qb, bf16_t* __restrict__ Kb,
                            const float* __restrict__ freqs) {
  size_t i = (size_t)blockIdx.x * blockDim.x + threadIdx.x;  // < 4096*16*64 = 2^22
  int dh = (int)(i & 63);
  int h  = (int)((i >> 6) & 15);
  size_t row = i >> 10;                   // 0..4095  (= b*2048 + si)
  int si = (int)(row & (S_LEN - 1));
  const float* fr = freqs + (size_t)si * HDIM;
  float c0 = __cosf(fr[dh]),      s0 = __sinf(fr[dh]);
  float c1 = __cosf(fr[dh + 64]), s1 = __sinf(fr[dh + 64]);
  size_t base = row * HID + (size_t)h * HDIM;
  float q0 = (float)Qb[base + dh], q1 = (float)Qb[base + dh + 64];
  Qb[base + dh]      = (bf16_t)(q0 * c0 - q1 * s0);
  Qb[base + dh + 64] = (bf16_t)(q1 * c1 + q0 * s1);
  float k0v = (float)Kb[base + dh], k1v = (float)Kb[base + dh + 64];
  Kb[base + dh]      = (bf16_t)(k0v * c0 - k1v * s0);
  Kb[base + dh + 64] = (bf16_t)(k1v * c1 + k0v * s1);
}

// ---------------------------------------------------------------------------
// V transpose: VT[(b*16+h)*128 + d][kpos] = V[b*2048+kpos][h*128+d]
// ---------------------------------------------------------------------------
__global__ void vtrans_kernel(const bf16_t* __restrict__ V, bf16_t* __restrict__ VT) {
  size_t i = (size_t)blockIdx.x * blockDim.x + threadIdx.x;  // < 2^23
  int kpos = (int)(i & (S_LEN - 1));
  int d    = (int)((i >> 11) & (HDIM - 1));
  int h    = (int)((i >> 18) & (NHEADS - 1));
  int b    = (int)(i >> 22);
  VT[i] = V[((size_t)(b * S_LEN + kpos)) * HID + (size_t)h * HDIM + d];
}

// ---------------------------------------------------------------------------
// Flash attention: block = (qtile of 128 rows, one (b,h)); 8 waves x 16 q-rows.
// ---------------------------------------------------------------------------
#define AKT 64             // kv tile width
#define KLD (HDIM + 8)     // 136 halfs = 272B row stride (16B aligned, no conflicts)
#define VLD (AKT + 8)      // 72 halfs = 144B
#define PLD (AKT + 8)      // 72 halfs

__global__ __launch_bounds__(256)
void attn_kernel(const bf16_t* __restrict__ Qb, const bf16_t* __restrict__ Kb,
                 const bf16_t* __restrict__ VT, const float* __restrict__ alibi,
                 const float* __restrict__ mask, bf16_t* __restrict__ O) {
  __shared__ __attribute__((aligned(16))) bf16_t sK[AKT * KLD];       // 17408B
  __shared__ __attribute__((aligned(16))) bf16_t sV[HDIM * VLD];      // 18432B
  __shared__ __attribute__((aligned(16))) bf16_t sP[8 * 16 * PLD];    // 18432B

  const int tid = threadIdx.x, wave = tid >> 5, lane = tid & 31;
  const int bh = blockIdx.y, b = bh >> 4, h = bh & 15;
  const int qrow0 = blockIdx.x * 128 + wave * 16;       // [0, 2048)
  const int coll = lane & 15, rtop = (lane >> 4) << 3;

  // Q fragments for this wave's 16 rows (resident across whole kv loop)
  const bf16_t* qbase = Qb + ((size_t)(b * S_LEN + qrow0)) * HID + (size_t)h * HDIM;
  v16bf qf[4];
#pragma unroll
  for (int dc = 0; dc < 4; ++dc) qf[dc] = load_frag(qbase + dc * 32, HID);

  const v8f zero8 = {0.f, 0.f, 0.f, 0.f, 0.f, 0.f, 0.f, 0.f};
  v8f oacc[8];
#pragma unroll
  for (int i = 0; i < 8; ++i) oacc[i] = zero8;
  float mrow[8], lrow[8];
#pragma unroll
  for (int r = 0; r < 8; ++r) { mrow[r] = -1e30f; lrow[r] = 0.0f; }

  const float inv_sqrt_d = 0.08838834764831845f;  // 1/sqrt(128)
  const float* alibi_h = alibi + (size_t)h * S_LEN;
  const bf16_t* kbase  = Kb + (size_t)b * S_LEN * HID + (size_t)h * HDIM;
  const bf16_t* vtbase = VT + (size_t)bh * HDIM * S_LEN;
  bf16_t* pw = sP + wave * 16 * PLD;

  for (int k0 = 0; k0 < S_LEN; k0 += AKT) {
    __syncthreads();
    // Async-stage K tile (64 rows x 128 halfs) and V^T tile (128 rows x 64 halfs)
#pragma unroll
    for (int s = 0; s < 4; ++s) {
      int seg = tid + s * 256;                 // 0..1023
      int row = seg >> 4, c = (seg & 15) << 3;
      async_copy16(kbase + (size_t)(k0 + row) * HID + c, sK + row * KLD + c);
    }
#pragma unroll
    for (int s = 0; s < 4; ++s) {
      int seg = tid + s * 256;
      int row = seg >> 3, c = (seg & 7) << 3;
      async_copy16(vtbase + (size_t)row * S_LEN + k0 + c, sV + row * VLD + c);
    }
    wait_async();
    __syncthreads();

    // scores: 16x64 per wave = 4 n-frags, each sum of 4 d-chunk WMMAs
    v8f sacc[4];
#pragma unroll
    for (int j = 0; j < 4; ++j) sacc[j] = zero8;
#pragma unroll
    for (int j = 0; j < 4; ++j)
#pragma unroll
      for (int dc = 0; dc < 4; ++dc)
        sacc[j] = wmma_bf16(qf[dc], load_frag(sK + (j * 16) * KLD + dc * 32, KLD),
                            sacc[j]);

    // scale + mask + alibi (mask is [1,1,s,s] broadcast over b)
#pragma unroll
    for (int j = 0; j < 4; ++j) {
      int col = k0 + j * 16 + coll;
      float al = alibi_h[col];
#pragma unroll
      for (int r = 0; r < 8; ++r) {
        int qrow = qrow0 + rtop + r;
        sacc[j][r] = sacc[j][r] * inv_sqrt_d + mask[(size_t)qrow * S_LEN + col] + al;
      }
    }

    // online softmax (row stats replicated across each 16-lane half)
#pragma unroll
    for (int r = 0; r < 8; ++r) {
      float mx = sacc[0][r];
#pragma unroll
      for (int j = 1; j < 4; ++j) mx = fmaxf(mx, sacc[j][r]);
#pragma unroll
      for (int off = 1; off < 16; off <<= 1) mx = fmaxf(mx, __shfl_xor(mx, off, 32));
      float mnew = fmaxf(mrow[r], mx);
      float corr = __expf(mrow[r] - mnew);
      mrow[r] = mnew;
      float psum = 0.0f;
#pragma unroll
      for (int j = 0; j < 4; ++j) {
        float p = __expf(sacc[j][r] - mnew);
        sacc[j][r] = p;
        psum += p;
      }
#pragma unroll
      for (int off = 1; off < 16; off <<= 1) psum += __shfl_xor(psum, off, 32);
      lrow[r] = lrow[r] * corr + psum;
#pragma unroll
      for (int i = 0; i < 8; ++i) oacc[i][r] *= corr;
    }

    // P (C-layout regs) -> wave-private LDS as bf16, reload as A-fragments
#pragma unroll
    for (int j = 0; j < 4; ++j)
#pragma unroll
      for (int r = 0; r < 8; ++r)
        pw[(rtop + r) * PLD + j * 16 + coll] = (bf16_t)sacc[j][r];
    __syncthreads();

    v16bf pf0 = load_frag(pw, PLD);        // k = 0..31 of this kv tile
    v16bf pf1 = load_frag(pw + 32, PLD);   // k = 32..63
#pragma unroll
    for (int i = 0; i < 8; ++i) {
      oacc[i] = wmma_bf16(pf0, load_frag(sV + (i * 16) * VLD, VLD), oacc[i]);
      oacc[i] = wmma_bf16(pf1, load_frag(sV + (i * 16) * VLD + 32, VLD), oacc[i]);
    }
  }

  // normalize and store to [b, s, H*d] (A-operand layout for the final GEMM)
  bf16_t* obase = O + ((size_t)(b * S_LEN + qrow0)) * HID + (size_t)h * HDIM;
#pragma unroll
  for (int r = 0; r < 8; ++r) {
    float inv = 1.0f / lrow[r];
#pragma unroll
    for (int i = 0; i < 8; ++i)
      obase[(size_t)(rtop + r) * HID + i * 16 + coll] = (bf16_t)(oacc[i][r] * inv);
  }
}

// ---------------------------------------------------------------------------
// Host-side orchestration
// ---------------------------------------------------------------------------
extern "C" void kernel_launch(void* const* d_in, const int* in_sizes, int n_in,
                              void* d_out, int out_size, void* d_ws, size_t ws_size,
                              hipStream_t stream) {
  (void)in_sizes; (void)n_in; (void)out_size; (void)ws_size;
  const float* x     = (const float*)d_in[0];
  const float* mask  = (const float*)d_in[1];
  const float* alibi = (const float*)d_in[2];
  const float* freqs = (const float*)d_in[3];
  const float* Wq    = (const float*)d_in[4];
  const float* Wk    = (const float*)d_in[5];
  const float* Wv    = (const float*)d_in[6];
  const float* Wo    = (const float*)d_in[7];

  char* ws = (char*)d_ws;
  const size_t actBytes = (size_t)MROWS * HID * sizeof(bf16_t);  // 16 MiB
  const size_t wBytes   = (size_t)HID * HID * sizeof(bf16_t);    //  8 MiB
  bf16_t* xb  = (bf16_t*)ws; ws += actBytes;
  bf16_t* wqb = (bf16_t*)ws; ws += wBytes;
  bf16_t* wkb = (bf16_t*)ws; ws += wBytes;
  bf16_t* wvb = (bf16_t*)ws; ws += wBytes;
  bf16_t* wob = (bf16_t*)ws; ws += wBytes;
  bf16_t* qb  = (bf16_t*)ws; ws += actBytes;
  bf16_t* kb  = (bf16_t*)ws; ws += actBytes;
  bf16_t* vb  = (bf16_t*)ws; ws += actBytes;
  bf16_t* vtb = (bf16_t*)ws; ws += actBytes;
  bf16_t* ob  = (bf16_t*)ws; ws += actBytes;

  const int nAct4 = MROWS * HID / 4;
  const int nW4   = HID * HID / 4;
  cvt_f32_to_bf16<<<(nAct4 + 255) / 256, 256, 0, stream>>>(x,  xb,  nAct4);
  cvt_f32_to_bf16<<<(nW4   + 255) / 256, 256, 0, stream>>>(Wq, wqb, nW4);
  cvt_f32_to_bf16<<<(nW4   + 255) / 256, 256, 0, stream>>>(Wk, wkb, nW4);
  cvt_f32_to_bf16<<<(nW4   + 255) / 256, 256, 0, stream>>>(Wv, wvb, nW4);
  cvt_f32_to_bf16<<<(nW4   + 255) / 256, 256, 0, stream>>>(Wo, wob, nW4);

  dim3 gg(HID / GTN, MROWS / GTM);  // (16, 32)
  gemm_bt<true ><<<gg, 256, 0, stream>>>(xb, wqb, nullptr, qb, MROWS, HID, HID);
  gemm_bt<true ><<<gg, 256, 0, stream>>>(xb, wkb, nullptr, kb, MROWS, HID, HID);
  gemm_bt<true ><<<gg, 256, 0, stream>>>(xb, wvb, nullptr, vb, MROWS, HID, HID);

  rope_kernel<<<(MROWS * NHEADS * 64) / 256, 256, 0, stream>>>(qb, kb, freqs);
  vtrans_kernel<<<(MROWS * HID) / 256, 256, 0, stream>>>(vb, vtb);

  attn_kernel<<<dim3(S_LEN / 128, BATCH * NHEADS), 256, 0, stream>>>(
      qb, kb, vtb, alibi, mask, ob);

  gemm_bt<false><<<gg, 256, 0, stream>>>(ob, wob, (float*)d_out, nullptr,
                                         MROWS, HID, HID);
}
